// Gemma3Attention_55542517071996
// MI455X (gfx1250) — matrix operand order repaired and checked by reading, not compiled, not run on hardware
//
#include <hip/hip_runtime.h>
#include <stdint.h>

#define S_LEN   4096
#define D_MODEL 1024
#define NH      8
#define DH      128
#define WIN     512

typedef __attribute__((ext_vector_type(16))) __bf16 v16bf;
typedef __attribute__((ext_vector_type(8)))  float  v8f;
typedef __attribute__((ext_vector_type(4)))  unsigned int v4u;
typedef __attribute__((ext_vector_type(8)))  int v8i;
typedef __attribute__((ext_vector_type(4)))  int v4i;

union Frag16 { uint32_t u[8]; v16bf v; };
union FragC  { float    f[8]; v8f   v; };

__device__ __forceinline__ unsigned short f2bf(float x) {
  union { float f; uint32_t u; } c; c.f = x;
  uint32_t r = c.u + 0x7FFFu + ((c.u >> 16) & 1u);
  return (unsigned short)(r >> 16);
}

__device__ __forceinline__ v8f wmma_bf16(const Frag16& a, const Frag16& b, v8f c) {
  return __builtin_amdgcn_wmma_f32_16x16x32_bf16(false, a.v, false, b.v,
                                                 (short)0, c, false, false);
}

// ---------------------------------------------------------------------------
// Tensor Data Mover: 2D tile (bf16) global -> LDS.
// D# layout per CDNA5 ISA 8.3/8.4: group0 {count=1, lds_addr, global_addr,
// type=2}; group1 {data_size=2B, tensor dims, tile dims, dim0 stride}.
// ---------------------------------------------------------------------------
__device__ __forceinline__ void tdm_load_2d_bf16(
    uint32_t lds_addr, const void* gaddr,
    uint32_t tile_d0, uint32_t tile_d1,
    uint32_t tens_d0, uint32_t tens_d1, uint32_t stride0) {
  uint64_t ga = (uint64_t)gaddr;
  v4u g0;
  g0.x = 1u;                                          // count=1 (valid), user
  g0.y = lds_addr;                                    // bits[63:32] LDS addr
  g0.z = (uint32_t)ga;                                // global addr lo
  g0.w = (uint32_t)((ga >> 32) & 0x1FFFFFFu) | (2u << 30); // addr hi | type=2
  v8i g1;
  g1[0] = (int)(1u << 16);                            // data_size=2B, mask=0
  g1[1] = (int)((tens_d0 & 0xFFFFu) << 16);           // tensor_dim0[15:0]
  g1[2] = (int)(((tens_d0 >> 16) & 0xFFFFu) |
                ((tens_d1 & 0xFFFFu) << 16));         // dim0[31:16]|dim1[15:0]
  g1[3] = (int)(((tens_d1 >> 16) & 0xFFFFu) |
                ((tile_d0 & 0xFFFFu) << 16));         // dim1[31:16]|tile0
  g1[4] = (int)(tile_d1 & 0xFFFFu);                   // tile1 | tile2=0
  g1[5] = (int)stride0;                               // dim0 stride [31:0]
  g1[6] = 0;
  g1[7] = 0;
  v4i z4; z4[0] = z4[1] = z4[2] = z4[3] = 0;
#if defined(__clang_major__) && (__clang_major__ >= 23)
  v8i z8; for (int i = 0; i < 8; ++i) z8[i] = 0;
  __builtin_amdgcn_tensor_load_to_lds(g0, g1, z4, z4, z8, 0);
#else
  __builtin_amdgcn_tensor_load_to_lds(g0, g1, z4, z4, 0);
#endif
}

// ---------------------------------------------------------------------------
// fp32 -> bf16 conversion (grid-stride)
// ---------------------------------------------------------------------------
__global__ void f32_to_bf16_kernel(const float* __restrict__ src,
                                   unsigned short* __restrict__ dst, int n) {
  int i = blockIdx.x * blockDim.x + threadIdx.x;
  int stride = gridDim.x * blockDim.x;
  for (; i < n; i += stride) dst[i] = f2bf(src[i]);
}

// ---------------------------------------------------------------------------
// C[M,N] = A[M,K] * B[N,K]^T   (bf16 in, fp32 out)
// block = 256 (8 waves); block tile 128x64; wave tile 16x64.
// B tile double-buffered in LDS via async-to-LDS DMA.
// All 4 B fragments are materialized in distinct registers before the 4
// WMMAs so the ds_loads coalesce into one clause and WMMAs issue back-to-back.
// ---------------------------------------------------------------------------
__global__ __launch_bounds__(256) void gemm_bf16_nt_kernel(
    const unsigned short* __restrict__ A,
    const unsigned short* __restrict__ B,
    float* __restrict__ C, int M, int N, int K) {
  __shared__ unsigned short bs[2][64 * 32];
  const int tid  = threadIdx.x;
  const int w    = tid >> 5;
  const int lane = tid & 31;
  const int l16  = lane & 15;
  const int half = lane >> 4;
  const int m0   = blockIdx.x * 128;
  const int n0   = blockIdx.y * 64;

  FragC acc[4];
#pragma unroll
  for (int t = 0; t < 4; ++t)
#pragma unroll
    for (int i = 0; i < 8; ++i) acc[t].f[i] = 0.f;

  const int brow = tid >> 2;        // 0..63
  const int bcol = (tid & 3) * 8;   // 0,8,16,24
  const uint32_t lds_off0 = (uint32_t)(size_t)&bs[0][brow * 32 + bcol];
  const uint32_t lds_off1 = (uint32_t)(size_t)&bs[1][brow * 32 + bcol];
  const unsigned short* Bbase = B + (size_t)n0 * K;
  const int arow = m0 + 16 * w + l16;

  // prologue: DMA first B tile into buffer 0
  {
    uint32_t voff = (uint32_t)(((brow * K) + bcol) * 2);
    asm volatile("global_load_async_to_lds_b128 %0, %1, %2"
                 :: "v"(lds_off0), "v"(voff), "s"(Bbase) : "memory");
  }

  int buf = 0;
  for (int k0 = 0; k0 < K; k0 += 32) {
    // issue next tile's DMA into the other buffer before computing this one
    if (k0 + 32 < K) {
      uint32_t voff = (uint32_t)(((brow * K) + (k0 + 32) + bcol) * 2);
      uint32_t ldsn = buf ? lds_off0 : lds_off1;
      asm volatile("global_load_async_to_lds_b128 %0, %1, %2"
                   :: "v"(ldsn), "v"(voff), "s"(Bbase) : "memory");
      asm volatile("s_wait_asynccnt 1" ::: "memory");   // older DMA done
    } else {
      asm volatile("s_wait_asynccnt 0" ::: "memory");
    }

    // A fragment (16x32 bf16, contiguous k = 16*half + s, matching B)
    Frag16 a;
#pragma unroll
    for (int v = 0; v < 8; ++v) {
      int kb = 16 * half + 2 * v;
      a.u[v] = *(const uint32_t*)&A[(size_t)arow * K + k0 + kb];
    }

    __syncthreads();   // current buffer visible to all waves

    const unsigned short* bsc = &bs[buf][0];
    Frag16 bfr[4];
#pragma unroll
    for (int t = 0; t < 4; ++t)
#pragma unroll
      for (int v = 0; v < 8; ++v) {
        int kk = 16 * half + 2 * v;
        int nr = t * 16 + l16;
        bfr[t].u[v] = (uint32_t)bsc[nr * 32 + kk] |
                      ((uint32_t)bsc[nr * 32 + kk + 1] << 16);
      }
#pragma unroll
    for (int t = 0; t < 4; ++t)
      acc[t].v = wmma_bf16(a, bfr[t], acc[t].v);

    __syncthreads();   // all waves done reading before buffer is re-DMAed
    buf ^= 1;
  }

#pragma unroll
  for (int t = 0; t < 4; ++t)
#pragma unroll
    for (int v = 0; v < 8; ++v) {
      int m = m0 + 16 * w + v + 8 * half;
      int n = n0 + t * 16 + l16;
      C[(size_t)m * N + n] = acc[t].f[v];
    }
}

// ---------------------------------------------------------------------------
// Per (s,h): RMSNorm(q,k) -> RoPE(q,k) -> bf16 Q/K [H][S][128], V^T [H][128][S]
// ---------------------------------------------------------------------------
__global__ __launch_bounds__(128) void rope_norm_kernel(
    const float* __restrict__ qf, const float* __restrict__ kf,
    const float* __restrict__ vf,
    const float* __restrict__ cosb, const float* __restrict__ sinb,
    const float* __restrict__ qw, const float* __restrict__ kw,
    unsigned short* __restrict__ Qb, unsigned short* __restrict__ Kb,
    unsigned short* __restrict__ Vt) {
  const int s = blockIdx.x;
  const int h = blockIdx.y;
  const int d = threadIdx.x;
  __shared__ float shq[DH], shk[DH], red[2][4];

  float q = qf[(size_t)s * D_MODEL + h * DH + d];
  float k = kf[(size_t)s * D_MODEL + h * DH + d];
  float v = vf[(size_t)s * D_MODEL + h * DH + d];

  float sq = q * q, sk = k * k;
#pragma unroll
  for (int off = 16; off >= 1; off >>= 1) {
    sq += __shfl_xor(sq, off, 32);
    sk += __shfl_xor(sk, off, 32);
  }
  const int wv = threadIdx.x >> 5;
  if ((threadIdx.x & 31) == 0) { red[0][wv] = sq; red[1][wv] = sk; }
  __syncthreads();
  sq = red[0][0] + red[0][1] + red[0][2] + red[0][3];
  sk = red[1][0] + red[1][1] + red[1][2] + red[1][3];

  float rq = rsqrtf(sq * (1.0f / DH) + 1e-6f);
  float rk = rsqrtf(sk * (1.0f / DH) + 1e-6f);
  float qn = q * rq * (1.0f + qw[d]);
  float kn = k * rk * (1.0f + kw[d]);
  shq[d] = qn; shk[d] = kn;
  __syncthreads();

  float c  = cosb[(size_t)s * DH + d];
  float si = sinb[(size_t)s * DH + d];
  float qo = (d < 64) ? -shq[d + 64] : shq[d - 64];
  float ko = (d < 64) ? -shk[d + 64] : shk[d - 64];
  float qr = qn * c + qo * si;
  float kr = kn * c + ko * si;

  Qb[((size_t)h * S_LEN + s) * DH + d] = f2bf(qr);
  Kb[((size_t)h * S_LEN + s) * DH + d] = f2bf(kr);
  Vt[((size_t)h * DH + d) * S_LEN + s] = f2bf(v);
}

// ---------------------------------------------------------------------------
// score tile from GLOBAL K: 16q x 16k, contraction over DH=128 via 4 WMMAs.
// B fragments batched before the WMMAs.
// ---------------------------------------------------------------------------
__device__ __forceinline__ void score_tile(const Frag16* aq,
                                           const unsigned short* Kh,
                                           int k0, int l16, int half, FragC& s) {
  Frag16 b[4];
#pragma unroll
  for (int d0i = 0; d0i < 4; ++d0i)
#pragma unroll
    for (int v = 0; v < 8; ++v) {
      int kk = 16 * half + 2 * v;
      b[d0i].u[v] =
          *(const uint32_t*)&Kh[(size_t)(k0 + l16) * DH + d0i * 32 + kk];
    }
#pragma unroll
  for (int d0i = 0; d0i < 4; ++d0i)
    s.v = wmma_bf16(aq[d0i], b[d0i], s.v);
}

// score tile from an LDS-staged 16x128 K tile (TDM destination)
__device__ __forceinline__ void score_tile_lds(const Frag16* aq,
                                               const unsigned short* kt,
                                               int l16, int half, FragC& s) {
  Frag16 b[4];
#pragma unroll
  for (int d0i = 0; d0i < 4; ++d0i)
#pragma unroll
    for (int v = 0; v < 8; ++v) {
      int kk = 16 * half + 2 * v;
      int idx = l16 * DH + d0i * 32 + kk;
      b[d0i].u[v] = (uint32_t)kt[idx] | ((uint32_t)kt[idx + 1] << 16);
    }
#pragma unroll
  for (int d0i = 0; d0i < 4; ++d0i)
    s.v = wmma_bf16(aq[d0i], b[d0i], s.v);
}

// ---------------------------------------------------------------------------
// Windowed flash attention with softcap; writes exact probs + bf16 attn out.
// 8 waves/block, each wave owns a 16-query tile; grid (S/128, H).
// Pass 1 streams K tiles through wave-private LDS via the Tensor Data Mover
// (double-buffered, TENSORcnt-synchronized).
// ---------------------------------------------------------------------------
__global__ __launch_bounds__(256) void attn_kernel(
    const unsigned short* __restrict__ Qb,
    const unsigned short* __restrict__ Kb,
    const unsigned short* __restrict__ Vt,
    float* __restrict__ probs,
    unsigned short* __restrict__ attn_b) {
  __shared__ unsigned short kt_lds[8][2][16 * DH];   // 64 KB: TDM K tiles
  __shared__ unsigned short p_lds[8][16 * 32];       // 8 KB: P routing
  const int tid  = threadIdx.x;
  const int w    = tid >> 5;
  const int lane = tid & 31;
  const int l16  = lane & 15;
  const int half = lane >> 4;
  const int h    = blockIdx.y;
  const int qt   = blockIdx.x * 128 + 16 * w;

  const unsigned short* Qh = Qb + (size_t)h * S_LEN * DH;
  const unsigned short* Kh = Kb + (size_t)h * S_LEN * DH;
  const unsigned short* Vh = Vt + (size_t)h * DH * S_LEN;

  // resident Q fragments (16x128 -> 4 A-fragments)
  Frag16 aq[4];
#pragma unroll
  for (int d0i = 0; d0i < 4; ++d0i)
#pragma unroll
    for (int v = 0; v < 8; ++v) {
      int kb = 16 * half + 2 * v;
      aq[d0i].u[v] =
          *(const uint32_t*)&Qh[(size_t)(qt + l16) * DH + d0i * 32 + kb];
    }

  const int lo = (qt >= WIN) ? (qt - WIN) : 0;
  const float kNegInf = -__builtin_inff();

  const uint32_t kbuf0 = (uint32_t)(size_t)&kt_lds[w][0][0];
  const uint32_t kbuf1 = (uint32_t)(size_t)&kt_lds[w][1][0];

  // ---- pass 1: streaming row max / sumexp, K tiles via TDM ---------------
  float mrun[8], lrun[8];
#pragma unroll
  for (int v = 0; v < 8; ++v) { mrun[v] = kNegInf; lrun[v] = 0.f; }

  // prologue TDM: first tile into buffer 0 (tile 128 cols x 16 rows)
  tdm_load_2d_bf16(kbuf0, Kh + (size_t)lo * DH, DH, 16, DH, S_LEN, DH);

  int bufi = 0;
  for (int k0 = lo; k0 <= qt; k0 += 16) {
    if (k0 + 16 <= qt) {
      tdm_load_2d_bf16(bufi ? kbuf0 : kbuf1, Kh + (size_t)(k0 + 16) * DH,
                       DH, 16, DH, S_LEN, DH);
      __builtin_amdgcn_s_wait_tensorcnt(1);   // current tile complete
    } else {
      __builtin_amdgcn_s_wait_tensorcnt(0);
    }
    asm volatile("" ::: "memory");

    FragC s;
#pragma unroll
    for (int v = 0; v < 8; ++v) s.f[v] = 0.f;
    score_tile_lds(aq, &kt_lds[w][bufi][0], l16, half, s);

#pragma unroll
    for (int v = 0; v < 8; ++v) {
      int q = qt + v + 8 * half;
      int key = k0 + l16;
      float sc = 50.f * tanhf(s.f[v] * (0.0625f / 50.f));
      if (key <= q && (q - key) < WIN) {
        float nm = fmaxf(mrun[v], sc);
        lrun[v] = lrun[v] * __expf(mrun[v] - nm) + __expf(sc - nm);
        mrun[v] = nm;
      }
    }
    bufi ^= 1;
  }
  // cross-lane merge within each 16-lane half
#pragma unroll
  for (int off = 8; off >= 1; off >>= 1) {
#pragma unroll
    for (int v = 0; v < 8; ++v) {
      float mo = __shfl_xor(mrun[v], off, 32);
      float lx = __shfl_xor(lrun[v], off, 32);
      float nm = fmaxf(mrun[v], mo);
      lrun[v] = lrun[v] * __expf(mrun[v] - nm) + lx * __expf(mo - nm);
      mrun[v] = nm;
    }
  }
  float rinv[8];
#pragma unroll
  for (int v = 0; v < 8; ++v) rinv[v] = 1.0f / lrun[v];

  // ---- pass 2: probs + P·V -----------------------------------------------
  FragC acc[8];
#pragma unroll
  for (int nt = 0; nt < 8; ++nt)
#pragma unroll
    for (int v = 0; v < 8; ++v) acc[nt].f[v] = 0.f;

  unsigned short* pl = p_lds[w];

  for (int kt = lo; kt <= qt; kt += 32) {
    if (kt + 32 <= qt)
      __builtin_prefetch(&Kh[(size_t)(kt + 32 + l16) * DH], 0, 0);
#pragma unroll
    for (int t = 0; t < 2; ++t) {
      int k0 = kt + 16 * t;
      if (k0 <= qt) {
        FragC s;
#pragma unroll
        for (int v = 0; v < 8; ++v) s.f[v] = 0.f;
        score_tile(aq, Kh, k0, l16, half, s);
#pragma unroll
        for (int v = 0; v < 8; ++v) {
          int q = qt + v + 8 * half;
          int key = k0 + l16;
          float sc = 50.f * tanhf(s.f[v] * (0.0625f / 50.f));
          bool valid = (key <= q) && ((q - key) < WIN);
          float p = valid ? __expf(sc - mrun[v]) * rinv[v] : 0.f;
          if (valid)
            probs[((size_t)h * S_LEN + q) * S_LEN + key] = p;
          pl[(v + 8 * half) * 32 + 16 * t + l16] = f2bf(p);
        }
      } else {
#pragma unroll
        for (int v = 0; v < 8; ++v)
          pl[(v + 8 * half) * 32 + 16 * t + l16] = 0;
      }
    }
    // P (16x32) as A fragment from wave-private LDS
    Frag16 pa;
#pragma unroll
    for (int v = 0; v < 8; ++v) {
      int kb = 16 * half + 2 * v;
      pa.u[v] = (uint32_t)pl[l16 * 32 + kb] |
                ((uint32_t)pl[l16 * 32 + kb + 1] << 16);
    }
    // V^T fragments: contraction over 32 keys, 8 d-tiles of 16,
    // batched 4 fragments at a time ahead of the WMMAs
#pragma unroll
    for (int g = 0; g < 2; ++g) {
      Frag16 bv[4];
#pragma unroll
      for (int j = 0; j < 4; ++j) {
        int nt = g * 4 + j;
#pragma unroll
        for (int v = 0; v < 8; ++v) {
          int kk = 16 * half + 2 * v;
          int key = kt + kk;
          if (key > S_LEN - 2) key = S_LEN - 2;  // p==0 there; safe clamp
          bv[j].u[v] =
              *(const uint32_t*)&Vh[(size_t)(nt * 16 + l16) * S_LEN + key];
        }
      }
#pragma unroll
      for (int j = 0; j < 4; ++j) {
        int nt = g * 4 + j;
        acc[nt].v = wmma_bf16(pa, bv[j], acc[nt].v);
      }
    }
  }

  // store attn in [s][h*128+d] layout (bf16) for the output projection GEMM
#pragma unroll
  for (int nt = 0; nt < 8; ++nt)
#pragma unroll
    for (int v = 0; v < 8; ++v) {
      int q = qt + v + 8 * half;
      int dcol = h * DH + nt * 16 + l16;
      attn_b[(size_t)q * D_MODEL + dcol] = f2bf(acc[nt].f[v]);
    }
}

// ---------------------------------------------------------------------------
extern "C" void kernel_launch(void* const* d_in, const int* in_sizes, int n_in,
                              void* d_out, int out_size, void* d_ws, size_t ws_size,
                              hipStream_t stream) {
  const float* hidden = (const float*)d_in[0];
  const float* cosb   = (const float*)d_in[1];
  const float* sinb   = (const float*)d_in[2];
  const float* Wq     = (const float*)d_in[3];
  const float* Wk     = (const float*)d_in[4];
  const float* Wv     = (const float*)d_in[5];
  const float* Wo     = (const float*)d_in[6];
  const float* qw     = (const float*)d_in[7];
  const float* kw     = (const float*)d_in[8];

  float* out   = (float*)d_out;
  float* probs = out + (size_t)S_LEN * D_MODEL;

  char* ws = (char*)d_ws;
  const size_t MB = 1u << 20;
  unsigned short* hb    = (unsigned short*)(ws);            // 8 MB
  unsigned short* wqb   = (unsigned short*)(ws +  8 * MB);  // 2 MB each
  unsigned short* wkb   = (unsigned short*)(ws + 10 * MB);
  unsigned short* wvb   = (unsigned short*)(ws + 12 * MB);
  unsigned short* wob   = (unsigned short*)(ws + 14 * MB);
  float*          qf    = (float*)(ws + 16 * MB);           // 16 MB each
  float*          kf    = (float*)(ws + 32 * MB);
  float*          vf    = (float*)(ws + 48 * MB);
  unsigned short* Qb    = (unsigned short*)(ws + 64 * MB);  // 8 MB each
  unsigned short* Kb    = (unsigned short*)(ws + 72 * MB);
  unsigned short* Vt    = (unsigned short*)(ws + 80 * MB);
  unsigned short* attnb = (unsigned short*)(ws + 88 * MB);

  // 1) fp32 -> bf16
  f32_to_bf16_kernel<<<1024, 256, 0, stream>>>(hidden, hb, S_LEN * D_MODEL);
  f32_to_bf16_kernel<<<512, 256, 0, stream>>>(Wq, wqb, D_MODEL * D_MODEL);
  f32_to_bf16_kernel<<<512, 256, 0, stream>>>(Wk, wkb, D_MODEL * D_MODEL);
  f32_to_bf16_kernel<<<512, 256, 0, stream>>>(Wv, wvb, D_MODEL * D_MODEL);
  f32_to_bf16_kernel<<<512, 256, 0, stream>>>(Wo, wob, D_MODEL * D_MODEL);

  // 2) QKV projections (WMMA GEMM, double-buffered async LDS staging)
  dim3 gg(S_LEN / 128, D_MODEL / 64);
  gemm_bf16_nt_kernel<<<gg, 256, 0, stream>>>(hb, wqb, qf, S_LEN, D_MODEL, D_MODEL);
  gemm_bf16_nt_kernel<<<gg, 256, 0, stream>>>(hb, wkb, kf, S_LEN, D_MODEL, D_MODEL);
  gemm_bf16_nt_kernel<<<gg, 256, 0, stream>>>(hb, wvb, vf, S_LEN, D_MODEL, D_MODEL);

  // 3) RMSNorm + RoPE, emit bf16 Q/K and transposed V
  rope_norm_kernel<<<dim3(S_LEN, NH), 128, 0, stream>>>(
      qf, kf, vf, cosb, sinb, qw, kw, Qb, Kb, Vt);

  // 4) probs is zero outside the window (exp(NEG - max) == 0)
  hipMemsetAsync(probs, 0, (size_t)NH * S_LEN * S_LEN * sizeof(float), stream);

  // 5) windowed attention (TDM-staged K tiles in pass 1)
  attn_kernel<<<dim3(S_LEN / 128, NH), 256, 0, stream>>>(Qb, Kb, Vt, probs, attnb);

  // 6) output projection
  gemm_bf16_nt_kernel<<<gg, 256, 0, stream>>>(attnb, wob, out, S_LEN, D_MODEL, D_MODEL);
}
